// MoE_2542620639786
// MI455X (gfx1250) — compile-verified, matrix operand order checked by reading, not run on hardware
//
#include <hip/hip_runtime.h>

// ---------------- problem constants (from reference setup) ----------------
#define BB 4096   // tokens
#define DD 1024   // model dim
#define HH 4096   // hidden dim
#define EE 8      // experts
#define LOSS_COEF 0.01f
#define EPSV 1e-10f

#define XROW 1032           // padded LDS row stride for x tile (bf16 elems), 2064B = 129*16B
#define HROW 136            // padded LDS row stride for h tile

// ---------------- WMMA types ----------------
typedef __attribute__((ext_vector_type(16))) __bf16 v16bf;
typedef __attribute__((ext_vector_type(8)))  __bf16 v8bf;
typedef __attribute__((ext_vector_type(8)))  float  v8f;

union Frag {
    v16bf v;
    v8bf  h[2];
};

// ---------------- init counters ----------------
__global__ void init_kernel(int* cnt) {
    if (threadIdx.x < EE) cnt[threadIdx.x] = 0;
}

// ---------------- fp32 -> bf16 elementwise (x) ----------------
__global__ void cvt_kernel(const float* __restrict__ in, __bf16* __restrict__ out, int n) {
    for (int i = blockIdx.x * blockDim.x + threadIdx.x; i < n; i += gridDim.x * blockDim.x)
        out[i] = (__bf16)in[i];
}

// ------- transpose+convert: in [E][R][C] f32 -> out [E][C][R] bf16 -------
__global__ void transpose_cvt_kernel(const float* __restrict__ in, __bf16* __restrict__ out,
                                     int R, int C) {
    __shared__ float t[32][33];
    int e = blockIdx.z;
    int c0 = blockIdx.x * 32, r0 = blockIdx.y * 32;
    const float* ip = in + (size_t)e * R * C;
    __bf16* op = out + (size_t)e * R * C;
    for (int i = threadIdx.y; i < 32; i += 8)
        t[i][threadIdx.x] = ip[(size_t)(r0 + i) * C + c0 + threadIdx.x];
    __syncthreads();
    for (int i = threadIdx.y; i < 32; i += 8)
        op[(size_t)(c0 + i) * R + r0 + threadIdx.x] = (__bf16)t[threadIdx.x][i];
}

// ---------------- gating: one wave per token ----------------
__global__ void gate_kernel(const float* __restrict__ x, const float* __restrict__ wg,
                            int* __restrict__ top_idx, float* __restrict__ top_gate,
                            int* __restrict__ gtok, int* __restrict__ gks,
                            int* __restrict__ cnt) {
    int lane = threadIdx.x & 31;
    int wv   = threadIdx.x >> 5;
    int b = blockIdx.x * 8 + wv;
    const float* xr = x + (size_t)b * DD;
    float acc[EE];
#pragma unroll
    for (int e = 0; e < EE; ++e) acc[e] = 0.f;
    for (int d = lane; d < DD; d += 32) {
        float xv = xr[d];
        const float* wr = wg + d * EE;
#pragma unroll
        for (int e = 0; e < EE; ++e) acc[e] += xv * wr[e];
    }
#pragma unroll
    for (int e = 0; e < EE; ++e) {
        float v = acc[e];
#pragma unroll
        for (int o = 16; o > 0; o >>= 1) v += __shfl_xor(v, o, 32);
        acc[e] = v;
    }
    if (lane == 0) {
        int i0 = 0; float m0 = acc[0];
#pragma unroll
        for (int e = 1; e < EE; ++e) if (acc[e] > m0) { m0 = acc[e]; i0 = e; }
        int i1 = (i0 == 0) ? 1 : 0; float m1 = acc[i1];
#pragma unroll
        for (int e = 0; e < EE; ++e) if (e != i0 && acc[e] > m1) { m1 = acc[e]; i1 = e; }
        float e1 = __expf(m1 - m0);
        float s = 1.f + e1;
        float g0 = 1.f / s, g1 = e1 / s;
        top_idx[2 * b]  = i0; top_idx[2 * b + 1]  = i1;
        top_gate[2 * b] = g0; top_gate[2 * b + 1] = g1;
        int p0 = atomicAdd(&cnt[i0], 1);
        gtok[i0 * BB + p0] = b; gks[i0 * BB + p0] = 0;
        int p1 = atomicAdd(&cnt[i1], 1);
        gtok[i1 * BB + p1] = b; gks[i1 * BB + p1] = 1;
    }
}

// ---------------- fused expert MLP: bf16 WMMA ----------------
// grid: (BB/32, EE), block: (32, 8).  Wave wv owns y columns [wv*128, wv*128+128).
// x tile (32 rows) is staged once in LDS and reused for every H chunk of layer 1.
// Per 128-wide H chunk: wave computes its 32x16 relu(h) block (K over D=1024),
// stages into LDS, then all waves consume it for layer 2 (K over H).
__global__ __launch_bounds__(256, 1)
void expert_kernel(const __bf16* __restrict__ xb, const __bf16* __restrict__ w1t,
                   const __bf16* __restrict__ w2t, const float* __restrict__ b1,
                   const float* __restrict__ b2, const int* __restrict__ gtok,
                   const int* __restrict__ gks, const int* __restrict__ cnt,
                   float* __restrict__ pair_out) {
    const int e = blockIdx.y;
    const int cnte = cnt[e];
    const int m0 = blockIdx.x * 32;
    if (m0 >= cnte) return;

    const int lane = threadIdx.x;      // 0..31
    const int wv   = threadIdx.y;      // 0..7
    const int l16  = lane & 15;
    const int lh   = lane >> 4;

    __shared__ alignas(16) __bf16 xs[32 * XROW];  // 66 KB staged x tile (padded rows)
    __shared__ alignas(16) __bf16 hs[32 * HROW];  // 8.7 KB h tile (padded rows)
    __shared__ int toks[32];
    __shared__ int kchs[32];

    if (wv == 0) {
        int mg = m0 + lane;
        toks[lane] = (mg < cnte) ? gtok[e * BB + mg] : gtok[e * BB];
        kchs[lane] = (mg < cnte) ? gks[e * BB + mg] : 0;
    }
    __syncthreads();

    // ---- stage gathered x rows into LDS (each thread copies 128 elems of one row) ----
    {
        int tid = wv * 32 + lane;
        int r  = tid >> 3;            // 0..31
        int c0 = (tid & 7) * 128;     // 0..896
        const v8bf* src = (const v8bf*)(xb + (size_t)toks[r] * DD + c0);
        v8bf* dst = (v8bf*)(xs + r * XROW + c0);
#pragma unroll
        for (int i = 0; i < 16; ++i) dst[i] = src[i];
    }
    __syncthreads();

    const __bf16* w1e = w1t + (size_t)e * HH * DD;   // [H][D]
    const __bf16* w2e = w2t + (size_t)e * DD * HH;   // [D][H]

    const v8f vzero = {0.f, 0.f, 0.f, 0.f, 0.f, 0.f, 0.f, 0.f};
    v8f acc[2][8];
#pragma unroll
    for (int ms = 0; ms < 2; ++ms)
#pragma unroll
        for (int nt = 0; nt < 8; ++nt) acc[ms][nt] = vzero;

    const int hcol = wv * 16 + l16;   // per-lane h column for layer 1
    const v8bf* arow0 = (const v8bf*)(xs + (0 * 16 + l16) * XROW);
    const v8bf* arow1 = (const v8bf*)(xs + (1 * 16 + l16) * XROW);

    for (int h0 = 0; h0 < HH; h0 += 128) {
        // ---------- layer 1 ----------
        const v8bf* brow1 = (const v8bf*)(w1e + (size_t)(h0 + hcol) * DD);
        v8f c1[2]; c1[0] = vzero; c1[1] = vzero;
#pragma unroll 4
        for (int k0 = 0; k0 < DD; k0 += 32) {
            int kb = k0 >> 3;
            Frag bf;
            bf.h[0] = brow1[kb + lh];
            bf.h[1] = brow1[kb + 2 + lh];
            Frag a0, a1;
            a0.h[0] = arow0[kb + lh];
            a0.h[1] = arow0[kb + 2 + lh];
            a1.h[0] = arow1[kb + lh];
            a1.h[1] = arow1[kb + 2 + lh];
            c1[0] = __builtin_amdgcn_wmma_f32_16x16x32_bf16(
                false, a0.v, false, bf.v, (short)0, c1[0], false, false);
            c1[1] = __builtin_amdgcn_wmma_f32_16x16x32_bf16(
                false, a1.v, false, bf.v, (short)0, c1[1], false, false);
        }
        float b1v = b1[e * HH + h0 + hcol];
#pragma unroll
        for (int ms = 0; ms < 2; ++ms) {
#pragma unroll
            for (int r = 0; r < 8; ++r) {
                float v = c1[ms][r] + b1v;
                v = v > 0.f ? v : 0.f;
                int mrow = ms * 16 + r + lh * 8;
                hs[mrow * HROW + hcol] = (__bf16)v;
            }
        }
        __syncthreads();

        // ---------- layer 2 ----------
#pragma unroll
        for (int kk = 0; kk < 128; kk += 32) {
            Frag a2[2];
#pragma unroll
            for (int ms = 0; ms < 2; ++ms) {
                const v8bf* hrow = (const v8bf*)(hs + (ms * 16 + l16) * HROW);
                a2[ms].h[0] = hrow[(kk >> 3) + lh];
                a2[ms].h[1] = hrow[(kk >> 3) + 2 + lh];
            }
#pragma unroll
            for (int nt = 0; nt < 8; ++nt) {
                int dcol = wv * 128 + nt * 16 + l16;
                const v8bf* brow2 = (const v8bf*)(w2e + (size_t)dcol * HH);
                Frag bf;
                bf.h[0] = brow2[((h0 + kk) >> 3) + lh];
                bf.h[1] = brow2[((h0 + kk) >> 3) + 2 + lh];
#pragma unroll
                for (int ms = 0; ms < 2; ++ms)
                    acc[ms][nt] = __builtin_amdgcn_wmma_f32_16x16x32_bf16(
                        false, a2[ms].v, false, bf.v, (short)0, acc[ms][nt], false, false);
            }
        }
        __syncthreads();
    }

    // ---------- write per-(token, k) result rows ----------
#pragma unroll
    for (int ms = 0; ms < 2; ++ms) {
#pragma unroll
        for (int nt = 0; nt < 8; ++nt) {
            int dcol = wv * 128 + nt * 16 + l16;
            float b2v = b2[e * DD + dcol];
#pragma unroll
            for (int r = 0; r < 8; ++r) {
                int mrow = ms * 16 + r + lh * 8;
                if (m0 + mrow < cnte) {
                    int tok = toks[mrow];
                    int kch = kchs[mrow];
                    pair_out[((size_t)tok * 2 + kch) * DD + dcol] = acc[ms][nt][r] + b2v;
                }
            }
        }
    }
}

// ---------------- deterministic combine: y = g0*p0 + g1*p1 ----------------
__global__ void combine_kernel(const float* __restrict__ pair_out,
                               const float* __restrict__ top_gate,
                               float* __restrict__ y) {
    int n = BB * DD;
    for (int i = blockIdx.x * blockDim.x + threadIdx.x; i < n; i += gridDim.x * blockDim.x) {
        int b = i >> 10;          // / DD
        int d = i & (DD - 1);
        float g0 = top_gate[2 * b], g1 = top_gate[2 * b + 1];
        y[i] = g0 * pair_out[((size_t)2 * b) * DD + d] +
               g1 * pair_out[((size_t)2 * b + 1) * DD + d];
    }
}

// ---------------- deterministic aux loss (single block) ----------------
__global__ void loss_kernel(const int* __restrict__ top_idx,
                            const float* __restrict__ top_gate,
                            float* __restrict__ out_loss) {
    __shared__ float red[256 * 16];
    int tid = threadIdx.x;
    float imp[EE], ld[EE];
#pragma unroll
    for (int e = 0; e < EE; ++e) { imp[e] = 0.f; ld[e] = 0.f; }
    for (int b = tid; b < BB; b += 256) {
        int i0 = top_idx[2 * b], i1 = top_idx[2 * b + 1];
        float g0 = top_gate[2 * b], g1 = top_gate[2 * b + 1];
#pragma unroll
        for (int e = 0; e < EE; ++e) {
            imp[e] += (i0 == e ? g0 : 0.f) + (i1 == e ? g1 : 0.f);
            ld[e]  += (i0 == e ? 1.f : 0.f) + ((i1 == e && g1 > 0.f) ? 1.f : 0.f);
        }
    }
#pragma unroll
    for (int e = 0; e < EE; ++e) { red[tid * 16 + e] = imp[e]; red[tid * 16 + 8 + e] = ld[e]; }
    __syncthreads();
    for (int s = 128; s > 0; s >>= 1) {
        if (tid < s)
#pragma unroll
            for (int j = 0; j < 16; ++j) red[tid * 16 + j] += red[(tid + s) * 16 + j];
        __syncthreads();
    }
    if (tid == 0) {
        float loss = 0.f;
#pragma unroll
        for (int which = 0; which < 2; ++which) {
            float mean = 0.f;
#pragma unroll
            for (int e = 0; e < EE; ++e) mean += red[which * 8 + e];
            mean *= (1.f / EE);
            float var = 0.f;
#pragma unroll
            for (int e = 0; e < EE; ++e) {
                float dlt = red[which * 8 + e] - mean;
                var += dlt * dlt;
            }
            var *= (1.f / (EE - 1));          // ddof = 1
            loss += var / (mean * mean + EPSV);
        }
        out_loss[0] = loss * LOSS_COEF;
    }
}

// ---------------- host launcher ----------------
extern "C" void kernel_launch(void* const* d_in, const int* in_sizes, int n_in,
                              void* d_out, int out_size, void* d_ws, size_t ws_size,
                              hipStream_t stream) {
    const float* x      = (const float*)d_in[0];   // [B, D]
    const float* w_gate = (const float*)d_in[1];   // [D, E]
    const float* W1     = (const float*)d_in[2];   // [E, D, H]
    const float* b1     = (const float*)d_in[3];   // [E, H]
    const float* W2     = (const float*)d_in[4];   // [E, H, D]
    const float* b2     = (const float*)d_in[5];   // [E, D]
    float* y = (float*)d_out;                      // [B, D] floats + 1 loss float

    // workspace carve-up (256B aligned regions)
    char* p = (char*)d_ws;
    size_t off = 0;
    auto carve = [&](size_t bytes) {
        void* r = p + off;
        off = (off + bytes + 255) & ~(size_t)255;
        return r;
    };
    __bf16* xb       = (__bf16*)carve((size_t)BB * DD * 2);            //  8 MB
    __bf16* w1t      = (__bf16*)carve((size_t)EE * HH * DD * 2);       // 64 MB [E][H][D]
    __bf16* w2t      = (__bf16*)carve((size_t)EE * DD * HH * 2);       // 64 MB [E][D][H]
    float*  pair_out = (float*) carve((size_t)BB * 2 * DD * 4);        // 32 MB
    int*    top_idx  = (int*)   carve((size_t)BB * 2 * 4);
    float*  top_gate = (float*) carve((size_t)BB * 2 * 4);
    int*    gtok     = (int*)   carve((size_t)EE * BB * 4);
    int*    gks      = (int*)   carve((size_t)EE * BB * 4);
    int*    cnt      = (int*)   carve(EE * 4);
    (void)ws_size; (void)in_sizes; (void)n_in; (void)out_size;

    init_kernel<<<1, 32, 0, stream>>>(cnt);
    cvt_kernel<<<2048, 256, 0, stream>>>(x, xb, BB * DD);
    // W1 [E][D][H] -> w1t [E][H][D]
    transpose_cvt_kernel<<<dim3(HH / 32, DD / 32, EE), dim3(32, 8), 0, stream>>>(W1, w1t, DD, HH);
    // W2 [E][H][D] -> w2t [E][D][H]
    transpose_cvt_kernel<<<dim3(DD / 32, HH / 32, EE), dim3(32, 8), 0, stream>>>(W2, w2t, HH, DD);
    gate_kernel<<<BB / 8, 256, 0, stream>>>(x, w_gate, top_idx, top_gate, gtok, gks, cnt);
    expert_kernel<<<dim3(BB / 32, EE), dim3(32, 8), 0, stream>>>(
        xb, w1t, w2t, b1, b2, gtok, gks, cnt, pair_out);
    loss_kernel<<<1, 256, 0, stream>>>(top_idx, top_gate, y + (size_t)BB * DD);
    combine_kernel<<<2048, 256, 0, stream>>>(pair_out, top_gate, y);
}
